// CounterPropagationNetwork_57999238365629
// MI455X (gfx1250) — matrix operand order repaired and checked by reading, not compiled, not run on hardware
//
#include <hip/hip_runtime.h>
#include <hip/hip_bf16.h>
#include <stdint.h>

// ---------------------------------------------------------------------------
// Problem sizes (compile-time constants from the reference)
// ---------------------------------------------------------------------------
#define BATCH   16384
#define KDIM    512      // INPUT_SIZE
#define HID     4096     // HIDDEN_SIZE
#define ODIM    512      // OUTPUT_SIZE

// Main-kernel tiling: 8 waves x (16 rows x 64 cols) per workgroup,
// double-buffered async K-tile staging.
#define M_WG    128      // batch rows per workgroup
#define NWAVES  8
#define BLOCK   (NWAVES * 32)
#define NT      64       // hidden columns per K chunk (4 subtiles per wave)
#define NCHUNK  (HID / NT)
#define XPITCH  520      // padded row pitch (ushorts): 1040B -> 4-bank row skew

typedef __attribute__((ext_vector_type(16))) __bf16 v16bf;
typedef __attribute__((ext_vector_type(8)))  float  v8f;
typedef __attribute__((ext_vector_type(4)))  float  f32x4;
typedef __attribute__((ext_vector_type(2)))  uint32_t u32x2;

union BFrag {
    v16bf v;
    f32x4 f[2];   // f[0] = bf16 elements 0..7, f[1] = 8..15
};

__device__ __forceinline__ unsigned short f2bf(float x) {
    unsigned int u = __builtin_bit_cast(unsigned int, x);
    unsigned int r = u + 0x7FFFu + ((u >> 16) & 1u);   // round-to-nearest-even
    return (unsigned short)(r >> 16);
}

// ---------------------------------------------------------------------------
// Kernel 1: per hidden row h: w_sq[h] = sum(K[h,:]^2), and bf16 copy of K.
// grid = HID blocks, 128 threads.
// ---------------------------------------------------------------------------
__global__ __launch_bounds__(128) void cpn_prepK(const float* __restrict__ K,
                                                 unsigned short* __restrict__ Kb,
                                                 float* __restrict__ wsq) {
    const int h    = blockIdx.x;
    const int tid  = threadIdx.x;
    const int lane = tid & 31;
    const int wid  = tid >> 5;
    const float* row = K + (size_t)h * KDIM;

    float s = 0.0f;
    #pragma unroll
    for (int i = tid; i < KDIM; i += 128) {
        float v = row[i];
        s += v * v;
        Kb[(size_t)h * KDIM + i] = f2bf(v);
    }
    #pragma unroll
    for (int off = 16; off >= 1; off >>= 1)
        s += __shfl_xor(s, off, 32);

    __shared__ float part[4];
    if (lane == 0) part[wid] = s;
    __syncthreads();
    if (tid == 0) wsq[h] = part[0] + part[1] + part[2] + part[3];
}

// ---------------------------------------------------------------------------
// Kernel 2: transpose G [ODIM x HID] -> Gt [HID x ODIM]
// grid = (HID/32, ODIM/32), block = (32, 8)
// ---------------------------------------------------------------------------
__global__ __launch_bounds__(256) void cpn_transposeG(const float* __restrict__ G,
                                                      float* __restrict__ Gt) {
    __shared__ float tile[32][33];
    const int x0 = blockIdx.x * 32;   // hidden index
    const int y0 = blockIdx.y * 32;   // output index
    #pragma unroll
    for (int j = threadIdx.y; j < 32; j += 8)
        tile[j][threadIdx.x] = G[(size_t)(y0 + j) * HID + x0 + threadIdx.x];
    __syncthreads();
    #pragma unroll
    for (int j = threadIdx.y; j < 32; j += 8)
        Gt[(size_t)(x0 + j) * ODIM + y0 + threadIdx.x] = tile[threadIdx.x][j];
}

// ---------------------------------------------------------------------------
// Async global->LDS staging of one K tile (NT rows x 512 bf16 = 64KB) using
// GLOBAL_LOAD_ASYNC_TO_LDS_B128 (ASYNCcnt-tracked, no VGPR round-trip).
// GVS addressing: uniform SGPR base + per-lane 32-bit byte offset.
// LDS dest address = low 32 bits of the flat shared-aperture pointer.
// ---------------------------------------------------------------------------
__device__ __forceinline__ void async_stage_k(const unsigned short* __restrict__ Kb,
                                              int ht, unsigned short* dstLds, int tid) {
    const char* gbase = (const char*)(Kb + (size_t)ht * KDIM);
    const uint32_t ldsBase = (uint32_t)(uintptr_t)dstLds;
    #pragma unroll
    for (int i = 0; i < (NT * (KDIM / 8)) / BLOCK; ++i) {   // 16 x b128 per thread
        const int idx = tid + i * BLOCK;
        const int r = idx >> 6;             // 64 x 16B per row
        const int c = idx & 63;
        uint32_t lds  = ldsBase + (uint32_t)(r * (XPITCH * 2) + c * 16);
        uint32_t goff = (uint32_t)(r * (KDIM * 2) + c * 16);
        asm volatile("global_load_async_to_lds_b128 %0, %1, %2 offset:0"
                     :: "v"(lds), "v"(goff), "s"(gbase)
                     : "memory");
    }
}

__device__ __forceinline__ void wait_async_zero() {
    asm volatile("s_wait_asynccnt 0x0" ::: "memory");
}

// ---------------------------------------------------------------------------
// Kernel 3: for each batch row, argmin_h ( wsq[h] - 2 * x.K[h] ) via WMMA.
// Each wave computes a 16x64 score block per chunk (A fragment reused across
// 4 B subtiles -> 2.5 ds_load_b128 per WMMA); K tiles are double-buffered and
// staged with async global->LDS copies overlapped with WMMA compute.
// grid = BATCH/M_WG blocks, BLOCK threads (8 waves).
// ---------------------------------------------------------------------------
__global__ __launch_bounds__(BLOCK) void cpn_argmin(const float* __restrict__ x,
                                                    const unsigned short* __restrict__ Kb,
                                                    const float* __restrict__ wsq,
                                                    int* __restrict__ winners) {
    __shared__ __align__(16) unsigned short xs[M_WG * XPITCH];    // 130 KB
    __shared__ __align__(16) unsigned short ks[2][NT * XPITCH];   // 2 x 65 KB
    __shared__ float wsqs[HID];                                   // 16 KB

    const int tid  = threadIdx.x;
    const int lane = tid & 31;
    const int wid  = tid >> 5;
    const int rowBase = blockIdx.x * M_WG;

    // ---- kick off async staging of K chunk 0 ----
    async_stage_k(Kb, 0, &ks[0][0], tid);

    // ---- stage w_sq ----
    #pragma unroll
    for (int i = tid; i < HID; i += BLOCK) wsqs[i] = wsq[i];

    // ---- stage x rows -> bf16 LDS (float4 global loads, uint2 LDS stores) ----
    {
        u32x2* xsu2 = (u32x2*)xs;                      // pitch 130 u32x2 per row
        const f32x4* xg = (const f32x4*)(x + (size_t)rowBase * KDIM);
        #pragma unroll 4
        for (int idx = tid; idx < M_WG * (KDIM / 4); idx += BLOCK) {
            const int r  = idx >> 7;          // /128 float4s per row
            const int c4 = idx & 127;
            f32x4 v = xg[(size_t)r * (KDIM / 4) + c4];
            u32x2 p;
            p.x = (uint32_t)f2bf(v.x) | ((uint32_t)f2bf(v.y) << 16);
            p.y = (uint32_t)f2bf(v.z) | ((uint32_t)f2bf(v.w) << 16);
            xsu2[r * (XPITCH / 4) + c4] = p;
        }
    }
    wait_async_zero();
    __syncthreads();

    // per-lane running min: element e holds row (wid*16 + 8*(lane>=16) + e),
    // candidate column h = ht + j*16 + (lane & 15)
    float minv[8];
    int   mini[8];
    #pragma unroll
    for (int e = 0; e < 8; ++e) { minv[e] = __builtin_huge_valf(); mini[e] = 0; }

    const int m0   = wid * 16;
    const int lnlo = lane & 15;
    const int aSel = (lane < 16) ? 0 : 8;    // A: 16x32 bf16 lane layout
    const int bSel = (lane < 16) ? 0 : 8;    // B: 32x16 bf16 lane layout
    const int aRow = (m0 + lnlo) * XPITCH;

    for (int c = 0; c < NCHUNK; ++c) {
        const int ht = c * NT;
        const unsigned short* kbuf = &ks[c & 1][0];

        // prefetch next chunk into the other buffer (overlaps with WMMAs below)
        if (c + 1 < NCHUNK)
            async_stage_k(Kb, ht + NT, &ks[(c + 1) & 1][0], tid);

        v8f acc[4];
        #pragma unroll
        for (int j = 0; j < 4; ++j) acc[j] = (v8f){};

        #pragma unroll
        for (int kb = 0; kb < KDIM; kb += 32) {
            BFrag a;
            a.f[0] = *(const f32x4*)&xs[aRow + kb + aSel];
            a.f[1] = *(const f32x4*)&xs[aRow + kb + 16 + aSel];
            #pragma unroll
            for (int j = 0; j < 4; ++j) {
                BFrag b;
                const int bRow = (j * 16 + lnlo) * XPITCH;
                b.f[0] = *(const f32x4*)&kbuf[bRow + kb + bSel];
                b.f[1] = *(const f32x4*)&kbuf[bRow + kb + bSel + 16];
                acc[j] = __builtin_amdgcn_wmma_f32_16x16x32_bf16(
                             false, a.v, false, b.v, (short)0, acc[j], false, false);
            }
        }

        #pragma unroll
        for (int j = 0; j < 4; ++j) {
            const int   h0   = ht + j * 16 + lnlo;
            const float wsqv = wsqs[h0];
            #pragma unroll
            for (int e = 0; e < 8; ++e) {
                float s = fmaf(-2.0f, acc[j][e], wsqv);
                bool better = s < minv[e];
                minv[e] = better ? s  : minv[e];
                mini[e] = better ? h0 : mini[e];
            }
        }

        wait_async_zero();   // own async copies for chunk c+1 are in LDS
        __syncthreads();     // everyone done reading ks[c&1] / writing ks[(c+1)&1]
    }

    // ---- reduce argmin across the 16 lanes sharing each row ----
    #pragma unroll
    for (int e = 0; e < 8; ++e) {
        float v = minv[e]; int idx = mini[e];
        #pragma unroll
        for (int off = 8; off >= 1; off >>= 1) {
            float ov = __shfl_xor(v,   off, 32);
            int   oi = __shfl_xor(idx, off, 32);
            if (ov < v || (ov == v && oi < idx)) { v = ov; idx = oi; }
        }
        minv[e] = v; mini[e] = idx;
    }
    if (lnlo == 0) {
        const int rb = rowBase + m0 + ((lane >> 4) << 3);
        #pragma unroll
        for (int e = 0; e < 8; ++e) winners[rb + e] = mini[e];
    }
}

// ---------------------------------------------------------------------------
// Kernel 4: out[b,:] = Gt[winner[b],:]   (float4 copy)
// grid = BATCH blocks, 128 threads (one float4 per thread).
// ---------------------------------------------------------------------------
__global__ __launch_bounds__(128) void cpn_gather(const float* __restrict__ Gt,
                                                  const int* __restrict__ winners,
                                                  float* __restrict__ out) {
    const int b = blockIdx.x;
    const int w = winners[b];
    const f32x4* src = (const f32x4*)(Gt + (size_t)w * ODIM);
    f32x4* dst = (f32x4*)(out + (size_t)b * ODIM);
    dst[threadIdx.x] = src[threadIdx.x];
}

// ---------------------------------------------------------------------------
extern "C" void kernel_launch(void* const* d_in, const int* in_sizes, int n_in,
                              void* d_out, int out_size, void* d_ws, size_t ws_size,
                              hipStream_t stream) {
    (void)in_sizes; (void)n_in; (void)out_size; (void)ws_size;
    const float* x  = (const float*)d_in[0];   // [BATCH, KDIM]
    const float* Kw = (const float*)d_in[1];   // [HID, KDIM]
    const float* Gw = (const float*)d_in[2];   // [ODIM, HID]

    float* out = (float*)d_out;                               // [BATCH, ODIM]
    int*   winners = (int*)(out + (size_t)BATCH * ODIM);      // [BATCH] (int bits)

    // workspace layout
    char* ws = (char*)d_ws;
    float*          Gt  = (float*)ws;                                     // 8 MB
    unsigned short* Kb  = (unsigned short*)(ws + (size_t)HID * ODIM * 4); // 4 MB
    float*          wsq = (float*)(ws + (size_t)HID * ODIM * 4
                                      + (size_t)HID * KDIM * 2);          // 16 KB

    cpn_prepK<<<HID, 128, 0, stream>>>(Kw, Kb, wsq);
    dim3 tg(HID / 32, ODIM / 32);
    cpn_transposeG<<<tg, dim3(32, 8), 0, stream>>>(Gw, Gt);
    cpn_argmin<<<BATCH / M_WG, BLOCK, 0, stream>>>(x, Kb, wsq, winners);
    cpn_gather<<<BATCH, 128, 0, stream>>>(Gt, winners, out);
}
